// Model_79826262163489
// MI455X (gfx1250) — matrix operand order repaired and checked by reading, not compiled, not run on hardware
//
#include <hip/hip_runtime.h>
#include <math.h>

// ---------------- model constants ----------------
static constexpr int Bc = 4, Sc = 512, Hc = 768, Lc = 12, NHc = 12, DHc = 64;
static constexpr int FFc = 3072, Cc = 9;
static constexpr int Pc = 512;              // 2 * att_span (BUCKETS=256)
static constexpr int Tc = Bc * Sc;          // 2048 tokens
static constexpr int BHc = Bc * NHc;        // 48 (b,h) pairs

typedef __bf16 bf16;
typedef __attribute__((ext_vector_type(16))) __bf16 v16bf;
typedef __attribute__((ext_vector_type(8)))  float  v8f;

#define USE_ASYNC_LDS 1

// ---- CDNA5 async global->LDS staging (ASYNCcnt path, ISA 08 §4) ----
__device__ __forceinline__ void stage_b128(bf16* ldst, const bf16* gsrc) {
#if USE_ASYNC_LDS
    // generic pointer to __shared__ lives in the LDS aperture; addr[31:0] is the LDS byte offset
    unsigned loff = (unsigned)(unsigned long long)ldst;
    asm volatile("global_load_async_to_lds_b128 %0, %1, off"
                 :: "v"(loff), "v"(gsrc) : "memory");
#else
    *reinterpret_cast<uint4*>(ldst) = *reinterpret_cast<const uint4*>(gsrc);
#endif
}
__device__ __forceinline__ void wait_stage() {
#if USE_ASYNC_LDS
    asm volatile("s_wait_asynccnt 0x0" ::: "memory");
#endif
}

// ---------------- embedding gather ----------------
__global__ __launch_bounds__(256)
void embed_kernel(const int* __restrict__ ids, const float* __restrict__ word_emb,
                  const float* __restrict__ tok_emb, float* __restrict__ out) {
    int t = blockIdx.x;
    long long id = ids[t];
    for (int i = threadIdx.x; i < Hc; i += 256)
        out[(long long)t * Hc + i] = word_emb[id * Hc + i] + tok_emb[i];
}

// ---------------- layernorm (optional residual add, mask, bf16 out) ----------------
__global__ __launch_bounds__(256)
void ln_kernel(const float* __restrict__ inA, const float* __restrict__ inB,
               const float* __restrict__ sc, const float* __restrict__ bi,
               const int* __restrict__ mask, float* __restrict__ outF,
               bf16* __restrict__ outB, int Hd) {
    int t = blockIdx.x, tid = threadIdx.x;
    __shared__ float rowv[768];
    __shared__ float red[256];
    const float* a = inA + (long long)t * Hd;
    const float* b = inB ? inB + (long long)t * Hd : nullptr;
    float lsum = 0.f;
    for (int i = tid; i < Hd; i += 256) {
        float v = a[i] + (b ? b[i] : 0.f);
        rowv[i] = v; lsum += v;
    }
    red[tid] = lsum; __syncthreads();
    for (int off = 128; off > 0; off >>= 1) {
        if (tid < off) red[tid] += red[tid + off];
        __syncthreads();
    }
    float mean = red[0] / (float)Hd;
    __syncthreads();
    float lvar = 0.f;
    for (int i = tid; i < Hd; i += 256) { float d = rowv[i] - mean; lvar += d * d; }
    red[tid] = lvar; __syncthreads();
    for (int off = 128; off > 0; off >>= 1) {
        if (tid < off) red[tid] += red[tid + off];
        __syncthreads();
    }
    float rs = rsqrtf(red[0] / (float)Hd + 1e-7f);
    float mf = mask ? (float)mask[t] : 1.0f;
    for (int i = tid; i < Hd; i += 256) {
        float o = ((rowv[i] - mean) * rs * sc[i] + bi[i]) * mf;
        if (outF) outF[(long long)t * Hd + i] = o;
        if (outB) outB[(long long)t * Hd + i] = (bf16)o;
    }
}

// ---------------- DeBERTa log-bucket relative position index tables ----------------
__global__ __launch_bounds__(256)
void rel_idx_kernel(int* __restrict__ c2pi, int* __restrict__ p2ci) {
    int idx = blockIdx.x * 256 + threadIdx.x;
    if (idx >= Sc * Sc) return;
    int q = idx / Sc, k = idx % Sc;
    int rel = q - k;
    const int mid = 128;                       // BUCKETS/2
    int sgn = (rel > 0) - (rel < 0);
    float abs_pos = (rel < mid && rel > -mid) ? (float)(mid - 1) : fabsf((float)rel);
    float log_pos = ceilf(logf(abs_pos / (float)mid) /
                          logf((512.0f - 1.0f) / (float)mid) * (float)(mid - 1)) + (float)mid;
    int bucket = (abs(rel) <= mid) ? rel : (int)(log_pos * (float)sgn);
    int c2 = bucket + 256; c2 = c2 < 0 ? 0 : (c2 > 511 ? 511 : c2);
    int p2 = -bucket + 256; p2 = p2 < 0 ? 0 : (p2 > 511 ? 511 : p2);
    c2pi[idx] = c2; p2ci[idx] = p2;
}

// ---------------- tiled transpose + f32->bf16 convert: dst[c*R+r] = src[r*Cd+c] ----------------
__global__ __launch_bounds__(256)
void transpose_bf16_kernel(const float* __restrict__ src, long long sZ0,
                           bf16* __restrict__ dst, long long dZ0, int R, int Cd) {
    __shared__ float tile[32][33];
    src += (long long)blockIdx.z * sZ0;
    dst += (long long)blockIdx.z * dZ0;
    int r0 = blockIdx.y * 32, c0 = blockIdx.x * 32;
    int tr = threadIdx.x >> 5, tc = threadIdx.x & 31;   // 8 x 32
#pragma unroll
    for (int i = 0; i < 32; i += 8)
        tile[tr + i][tc] = src[(long long)(r0 + tr + i) * Cd + c0 + tc];
    __syncthreads();
#pragma unroll
    for (int i = 0; i < 32; i += 8)
        dst[(long long)(c0 + tr + i) * R + r0 + tc] = (bf16)tile[tc][tr + i];
}

// ---------------- WMMA GEMM (NT): C[M,N] = A[M,K] * B[N,K]^T, tile 64x64, K-step 32 ----------------
// Fragment layouts (wave32; A/B contiguous k as used here, C/D per ISA 7.12.2):
//  A 16x32 bf16: lane m=lane&15; K = half*16 .. half*16+15
//  B 32x16 bf16: lane n=lane&15; K = half*16 .. half*16+15
//  C/D 16x16 f32: lane n=lane&15, VGPR r -> m = 8*(lane>>4)+r
static constexpr int LDSW = 40;  // padded row stride (bf16): 80B = 20 banks, 16B aligned

__device__ __forceinline__ void gemm_core(const bf16* ldsA, const bf16* ldsB,
                                          v8f acc[4], int lane, int wave) {
    int half = lane >> 4, l16 = lane & 15;
    union { v16bf v; uint4 q[2]; } af;
    int arow = wave * 16 + l16;
    af.q[0] = *reinterpret_cast<const uint4*>(&ldsA[arow * LDSW + half * 16]);
    af.q[1] = *reinterpret_cast<const uint4*>(&ldsA[arow * LDSW + half * 16 + 8]);
#pragma unroll
    for (int nt = 0; nt < 4; ++nt) {
        union { v16bf v; uint4 q[2]; } bfg;
        int brow = nt * 16 + l16;
        bfg.q[0] = *reinterpret_cast<const uint4*>(&ldsB[brow * LDSW + half * 16]);
        bfg.q[1] = *reinterpret_cast<const uint4*>(&ldsB[brow * LDSW + half * 16 + 8]);
        acc[nt] = __builtin_amdgcn_wmma_f32_16x16x32_bf16(
            false, af.v, false, bfg.v, (short)0, acc[nt], false, false);
    }
}

__global__ __launch_bounds__(128)
void gemm_nt(const bf16* __restrict__ A, int lda, long long aZ0, long long aZ1,
             const bf16* __restrict__ Bn, int ldb, long long bZ0, long long bZ1,
             float* __restrict__ C, int ldc, long long cZ0, long long cZ1,
             bf16* __restrict__ Cb, const float* __restrict__ bias,
             int M, int N, int K, int nz1) {
    __shared__ bf16 ldsA[64 * LDSW];
    __shared__ bf16 ldsB[64 * LDSW];
    long long z0 = blockIdx.z / nz1, z1 = blockIdx.z % nz1;
    A  += z0 * aZ0 + z1 * aZ1;
    Bn += z0 * bZ0 + z1 * bZ1;
    if (C)  C  += z0 * cZ0 + z1 * cZ1;
    if (Cb) Cb += z0 * cZ0 + z1 * cZ1;
    int m0 = blockIdx.y * 64, n0 = blockIdx.x * 64;
    int tid = threadIdx.x, lane = tid & 31, wave = tid >> 5;
    int row0 = tid >> 2, ch0 = (tid & 3) * 8;          // chunk 0: rows 0..31
    v8f acc[4] = {};
    for (int k0 = 0; k0 < K; k0 += 32) {
#pragma unroll
        for (int j = 0; j < 2; ++j) {                   // fixed trip count, no exec masking
            int row = row0 + j * 32, ch = ch0;
            stage_b128(&ldsA[row * LDSW + ch], &A[(long long)(m0 + row) * lda + k0 + ch]);
            stage_b128(&ldsB[row * LDSW + ch], &Bn[(long long)(n0 + row) * ldb + k0 + ch]);
        }
        wait_stage();
        __syncthreads();
        gemm_core(ldsA, ldsB, acc, lane, wave);
        __syncthreads();
    }
    int half = lane >> 4, l16 = lane & 15;
#pragma unroll
    for (int nt = 0; nt < 4; ++nt) {
        int col = n0 + nt * 16 + l16;
        float bv = bias ? bias[col] : 0.f;
#pragma unroll
        for (int r = 0; r < 8; ++r) {
            int row = m0 + wave * 16 + half * 8 + r;
            float val = acc[nt][r] + bv;
            if (C)  C[(long long)row * ldc + col] = val;
            if (Cb) Cb[(long long)row * ldc + col] = (bf16)val;
        }
    }
}

// ---------------- fused gather + mask + softmax over k ----------------
__global__ __launch_bounds__(256)
void attn_softmax(float* __restrict__ scores, bf16* __restrict__ probsb,
                  const float* __restrict__ c2p, const float* __restrict__ p2c,
                  const int* __restrict__ c2pi, const int* __restrict__ p2ci,
                  const int* __restrict__ mask, float scale) {
    int q = blockIdx.x, bh = blockIdx.y, b = bh / NHc, tid = threadIdx.x;
    float* row = scores + ((long long)bh * Sc + q) * Sc;
    const float* c2pr = c2p + ((long long)bh * Sc + q) * Pc;
    const float* p2cb = p2c + (long long)bh * Sc * Pc;
    int mq = mask[b * Sc + q];
    __shared__ float red[256];
    float vals[2]; float mx = -3.0e38f;
#pragma unroll
    for (int j = 0; j < 2; ++j) {
        int k = tid + j * 256;
        float s = row[k] + c2pr[c2pi[q * Sc + k]]
                         + p2cb[(long long)k * Pc + p2ci[k * Sc + q]];
        s *= scale;
        s = (mq && mask[b * Sc + k]) ? s : -3.0e38f;
        vals[j] = s; mx = fmaxf(mx, s);
    }
    red[tid] = mx; __syncthreads();
    for (int off = 128; off > 0; off >>= 1) {
        if (tid < off) red[tid] = fmaxf(red[tid], red[tid + off]);
        __syncthreads();
    }
    mx = red[0]; __syncthreads();
    float sum = 0.f;
#pragma unroll
    for (int j = 0; j < 2; ++j) { vals[j] = __expf(vals[j] - mx); sum += vals[j]; }
    red[tid] = sum; __syncthreads();
    for (int off = 128; off > 0; off >>= 1) {
        if (tid < off) red[tid] += red[tid + off];
        __syncthreads();
    }
    float inv = 1.0f / red[0];
#pragma unroll
    for (int j = 0; j < 2; ++j) {
        int k = tid + j * 256;
        float p = vals[j] * inv;
        row[k] = p;
        probsb[((long long)bh * Sc + q) * Sc + k] = (bf16)p;
    }
}

// ---------------- exact GELU -> bf16 ----------------
__global__ __launch_bounds__(256)
void gelu_kernel(const float* __restrict__ in, bf16* __restrict__ out, long long n) {
    long long i = (long long)blockIdx.x * 256 + threadIdx.x;
    if (i >= n) return;
    float v = in[i];
    out[i] = (bf16)(0.5f * v * (1.0f + erff(v * 0.70710678118654752f)));
}

// ---------------- classifier (N=9, plain dot) ----------------
__global__ __launch_bounds__(256)
void cls_kernel(const float* __restrict__ x, const float* __restrict__ w,
                const float* __restrict__ bias, float* __restrict__ out) {
    int idx = blockIdx.x * 256 + threadIdx.x;
    if (idx >= Tc * Cc) return;
    int t = idx / Cc, c = idx % Cc;
    float s = bias[c];
    const float* xr = x + (long long)t * Hc;
    for (int i = 0; i < Hc; ++i) s += xr[i] * w[i * Cc + c];
    out[idx] = s;
}

// ---------------- host orchestration ----------------
extern "C" void kernel_launch(void* const* d_in, const int* in_sizes, int n_in,
                              void* d_out, int out_size, void* d_ws, size_t ws_size,
                              hipStream_t stream) {
    (void)in_sizes; (void)n_in; (void)out_size; (void)ws_size;
    const int*   ids    = (const int*)  d_in[0];
    const int*   amask  = (const int*)  d_in[1];
    const float* wemb   = (const float*)d_in[2];
    const float* temb   = (const float*)d_in[3];
    const float* eln_s  = (const float*)d_in[4];
    const float* eln_b  = (const float*)d_in[5];
    const float* remb   = (const float*)d_in[6];
    const float* rln_s  = (const float*)d_in[7];
    const float* rln_b  = (const float*)d_in[8];
    const float* q_w    = (const float*)d_in[9];
    const float* q_b    = (const float*)d_in[10];
    const float* k_w    = (const float*)d_in[11];
    const float* k_b    = (const float*)d_in[12];
    const float* v_w    = (const float*)d_in[13];
    const float* v_b    = (const float*)d_in[14];
    const float* ao_w   = (const float*)d_in[15];
    const float* ao_b   = (const float*)d_in[16];
    const float* aln_s  = (const float*)d_in[17];
    const float* aln_b  = (const float*)d_in[18];
    const float* i_w    = (const float*)d_in[19];
    const float* i_b    = (const float*)d_in[20];
    const float* o_w    = (const float*)d_in[21];
    const float* o_b    = (const float*)d_in[22];
    const float* oln_s  = (const float*)d_in[23];
    const float* oln_b  = (const float*)d_in[24];
    const float* cls_w  = (const float*)d_in[25];
    const float* cls_b  = (const float*)d_in[26];
    float* out = (float*)d_out;

    size_t cur = 0;
    auto alloc = [&](size_t bytes) -> void* {
        void* p = (char*)d_ws + cur;
        cur += (bytes + 255) & ~(size_t)255;
        return p;
    };
    const size_t TH = (size_t)Tc * Hc;
    float* y      = (float*)alloc(TH * 4);                       // residual branch / embed tmp
    float* x      = (float*)alloc(TH * 4);
    bf16*  xb     = (bf16*) alloc(TH * 2);
    float* vf     = (float*)alloc(TH * 4);
    bf16*  qb     = (bf16*) alloc(TH * 2);
    bf16*  kb     = (bf16*) alloc(TH * 2);
    bf16*  vt     = (bf16*) alloc(TH * 2);                       // V transposed: [B][H][S]
    bf16*  ctxb   = (bf16*) alloc(TH * 2);
    bf16*  pkb    = (bf16*) alloc((size_t)Pc * Hc * 2);
    bf16*  pqb    = (bf16*) alloc((size_t)Pc * Hc * 2);
    bf16*  relb   = (bf16*) alloc((size_t)Pc * Hc * 2);
    float* scores = (float*)alloc((size_t)BHc * Sc * Sc * 4);
    bf16*  probsb = (bf16*) alloc((size_t)BHc * Sc * Sc * 2);
    float* c2p    = (float*)alloc((size_t)BHc * Sc * Pc * 4);
    float* p2c    = (float*)alloc((size_t)BHc * Sc * Pc * 4);
    float* hmid   = (float*)alloc((size_t)Tc * FFc * 4);
    bf16*  hmidb  = (bf16*) alloc((size_t)Tc * FFc * 2);
    int*   c2pi   = (int*)  alloc((size_t)Sc * Sc * 4);
    int*   p2ci   = (int*)  alloc((size_t)Sc * Sc * 4);
    bf16*  qwT    = (bf16*) alloc((size_t)Hc * Hc * 2);          // per-layer W^T bf16
    bf16*  kwT    = (bf16*) alloc((size_t)Hc * Hc * 2);
    bf16*  vwT    = (bf16*) alloc((size_t)Hc * Hc * 2);
    bf16*  aowT   = (bf16*) alloc((size_t)Hc * Hc * 2);
    bf16*  fwT    = (bf16*) alloc((size_t)Hc * FFc * 2);         // shared by iw then ow

    const float scale = 1.0f / sqrtf((float)DHc * 3.0f);

    embed_kernel<<<Tc, 256, 0, stream>>>(ids, wemb, temb, y);
    ln_kernel<<<Tc, 256, 0, stream>>>(y, nullptr, eln_s, eln_b, amask, x, xb, Hc);
    ln_kernel<<<Pc, 256, 0, stream>>>(remb, nullptr, rln_s, rln_b, nullptr, nullptr, relb, Hc);
    rel_idx_kernel<<<(Sc * Sc + 255) / 256, 256, 0, stream>>>(c2pi, p2ci);

    dim3 gProj(Hc / 64, Tc / 64, 1);          // 2048x768  K=768 (or K=3072 for FFN2)
    dim3 gPos (Hc / 64, Pc / 64, 1);          // 512x768   K=768
    dim3 gAttn(Sc / 64, Sc / 64, BHc);        // 512x512   K=64, batched (b,h)
    dim3 gCtx (1,       Sc / 64, BHc);        // 512x64    K=512
    dim3 gFF1 (FFc / 64, Tc / 64, 1);
    dim3 gTrH (Hc / 32, Hc / 32, 1);          // 768x768 weight transpose
    dim3 gTrI (FFc / 32, Hc / 32, 1);         // i_w [768,3072]
    dim3 gTrO (Hc / 32, FFc / 32, 1);         // o_w [3072,768]
    dim3 gTrV (Hc / 32, Sc / 32, Bc);         // V [S,H] per batch
    const long long SH = (long long)Sc * Hc, SS = (long long)Sc * Sc, SP = (long long)Sc * Pc;
    const long long HS = (long long)Hc * Sc;

    for (int l = 0; l < Lc; ++l) {
        const float* qbi = q_b + (size_t)l * Hc;
        const float* kbi = k_b + (size_t)l * Hc;
        const float* vbi = v_b + (size_t)l * Hc;

        // one-shot weight transpose+convert (f32 -> bf16 [N,K])
        transpose_bf16_kernel<<<gTrH, 256, 0, stream>>>(q_w + (size_t)l * Hc * Hc, 0, qwT, 0, Hc, Hc);
        transpose_bf16_kernel<<<gTrH, 256, 0, stream>>>(k_w + (size_t)l * Hc * Hc, 0, kwT, 0, Hc, Hc);
        transpose_bf16_kernel<<<gTrH, 256, 0, stream>>>(v_w + (size_t)l * Hc * Hc, 0, vwT, 0, Hc, Hc);
        transpose_bf16_kernel<<<gTrH, 256, 0, stream>>>(ao_w + (size_t)l * Hc * Hc, 0, aowT, 0, Hc, Hc);

        // Q/K/V projections; shared positional projections pk, pq
        gemm_nt<<<gProj, 128, 0, stream>>>(xb, Hc, 0, 0, qwT, Hc, 0, 0,
                                           nullptr, Hc, 0, 0, qb, qbi, Tc, Hc, Hc, 1);
        gemm_nt<<<gProj, 128, 0, stream>>>(xb, Hc, 0, 0, kwT, Hc, 0, 0,
                                           nullptr, Hc, 0, 0, kb, kbi, Tc, Hc, Hc, 1);
        gemm_nt<<<gProj, 128, 0, stream>>>(xb, Hc, 0, 0, vwT, Hc, 0, 0,
                                           vf, Hc, 0, 0, nullptr, vbi, Tc, Hc, Hc, 1);
        gemm_nt<<<gPos, 128, 0, stream>>>(relb, Hc, 0, 0, kwT, Hc, 0, 0,
                                          nullptr, Hc, 0, 0, pkb, kbi, Pc, Hc, Hc, 1);
        gemm_nt<<<gPos, 128, 0, stream>>>(relb, Hc, 0, 0, qwT, Hc, 0, 0,
                                          nullptr, Hc, 0, 0, pqb, qbi, Pc, Hc, Hc, 1);
        // V^T for probs@V
        transpose_bf16_kernel<<<gTrV, 256, 0, stream>>>(vf, SH, vt, HS, Sc, Hc);

        // content scores Q.K^T, c2p table Q.pk^T, p2c table K.pq^T (batched over b,h)
        gemm_nt<<<gAttn, 128, 0, stream>>>(qb, Hc, SH, 64, kb, Hc, SH, 64,
                                           scores, Sc, (long long)NHc * SS, SS,
                                           nullptr, nullptr, Sc, Sc, DHc, NHc);
        gemm_nt<<<gAttn, 128, 0, stream>>>(qb, Hc, SH, 64, pkb, Hc, 0, 64,
                                           c2p, Pc, (long long)NHc * SP, SP,
                                           nullptr, nullptr, Sc, Pc, DHc, NHc);
        gemm_nt<<<gAttn, 128, 0, stream>>>(kb, Hc, SH, 64, pqb, Hc, 0, 64,
                                           p2c, Pc, (long long)NHc * SP, SP,
                                           nullptr, nullptr, Sc, Pc, DHc, NHc);

        attn_softmax<<<dim3(Sc, BHc), 256, 0, stream>>>(scores, probsb, c2p, p2c,
                                                        c2pi, p2ci, amask, scale);

        // ctx = probs @ V  (B = V^T slice [64,S]), writes bf16 [b, q, h*64+d]
        gemm_nt<<<gCtx, 128, 0, stream>>>(probsb, Sc, (long long)NHc * SS, SS,
                                          vt, Sc, HS, 64LL * Sc,
                                          nullptr, Hc, SH, 64, ctxb, nullptr, Sc, DHc, Sc, NHc);
        // attention output projection + residual LN
        gemm_nt<<<gProj, 128, 0, stream>>>(ctxb, Hc, 0, 0, aowT, Hc, 0, 0,
                                           y, Hc, 0, 0, nullptr, ao_b + (size_t)l * Hc, Tc, Hc, Hc, 1);
        ln_kernel<<<Tc, 256, 0, stream>>>(x, y, aln_s + (size_t)l * Hc, aln_b + (size_t)l * Hc,
                                          nullptr, x, xb, Hc);
        // FFN
        transpose_bf16_kernel<<<gTrI, 256, 0, stream>>>(i_w + (size_t)l * Hc * FFc, 0, fwT, 0, Hc, FFc);
        gemm_nt<<<gFF1, 128, 0, stream>>>(xb, Hc, 0, 0, fwT, Hc, 0, 0,
                                          hmid, FFc, 0, 0, nullptr, i_b + (size_t)l * FFc,
                                          Tc, FFc, Hc, 1);
        gelu_kernel<<<(int)(((long long)Tc * FFc + 255) / 256), 256, 0, stream>>>(
            hmid, hmidb, (long long)Tc * FFc);
        transpose_bf16_kernel<<<gTrO, 256, 0, stream>>>(o_w + (size_t)l * FFc * Hc, 0, fwT, 0, FFc, Hc);
        gemm_nt<<<gProj, 128, 0, stream>>>(hmidb, FFc, 0, 0, fwT, FFc, 0, 0,
                                           y, Hc, 0, 0, nullptr, o_b + (size_t)l * Hc, Tc, Hc, FFc, 1);
        ln_kernel<<<Tc, 256, 0, stream>>>(x, y, oln_s + (size_t)l * Hc, oln_b + (size_t)l * Hc,
                                          nullptr, x, xb, Hc);
    }
    cls_kernel<<<(Tc * Cc + 255) / 256, 256, 0, stream>>>(x, cls_w, cls_b, out);
}
